// MambaMUNet_22797686407367
// MI455X (gfx1250) — matrix operand order repaired and checked by reading, not compiled, not run on hardware
//
#include <hip/hip_runtime.h>
#include <hip/hip_bf16.h>

// ---------------------------------------------------------------------------
// MambaMUNet for MI455X (gfx1250, wave32).
// Heavy GEMMs: V_WMMA_F32_16X16X32_BF16, B-strip staged in LDS via
// GLOBAL_LOAD_ASYNC_TO_LDS_B128 (double-buffered, ASYNCcnt-tracked).
// Scan / conv / epilogues are fp32 VALU (bandwidth-bound, not matmul shaped).
// ---------------------------------------------------------------------------

typedef __attribute__((ext_vector_type(16))) __bf16 v16bf;
typedef __attribute__((ext_vector_type(8)))  __bf16 v8bf;
typedef __attribute__((ext_vector_type(8)))  float  v8f;

namespace cfg {
constexpr int B      = 8;
constexpr int CIN    = 256;
constexpr int L      = 2048;
constexpr int DIN    = 512;     // EXPAND * CIN
constexpr int DSTATE = 16;
constexpr int DCONV  = 4;
constexpr int DTRANK = 16;
constexpr int DTK    = 32;      // dt_proj K padded to one WMMA K-step
constexpr int ML     = B * L;   // 16384 GEMM rows
constexpr int NSPK   = 2;
}

// ---------------------------------------------------------------------------
// WMMA GEMM: C[M,N] = A[M,K] * W[N,K]^T   (A, W bf16 row-major; f32 acc)
//
// Block = 256 thr = 8 waves; block tile 128M x 64N. Wave w owns M-tile
// (blockIdx.y*8+w) and all four 16-wide N-tiles of the block's N strip, so
// each A fragment feeds 4 WMMAs from registers. The 64Nx32K bf16 B strip
// (4 KB) is staged in LDS by async DMA: each thread moves one 16B slice with
// global_load_async_to_lds_b128; double buffered, s_wait_asynccnt + barrier.
//
// Fragment layouts (wave32):
//  A (16x32): lane<16 -> M=lane,   K = {k0..k0+7, k0+16..k0+23}
//             lane>=16 -> M=lane-16, K shifted by 8.
//  B (32x16): lane -> N=lane&15, K same as A (k0..k0+7, k0+16..k0+23; +8 for lanes>=16).
//  C/D: acc[r] -> M = 16*tileM + r + 8*(lane>=16), N = 16*tileN + (lane&15)
// ---------------------------------------------------------------------------
__global__ void __launch_bounds__(256)
wmma_gemm_bf16(const __bf16* __restrict__ A,
               const __bf16* __restrict__ W,
               float*  __restrict__ Cf,    // optional f32 out
               __bf16* __restrict__ Cb,    // optional bf16 out
               int N, int K)
{
    __shared__ __bf16 bsm[2 * 64 * 32];          // 2 x 4KB B-strip buffers

    const int t    = threadIdx.x;
    const int lane = t & 31;
    const int wave = t >> 5;
    const int hi   = lane >> 4;
    const int lidx = lane & 15;

    const int n0    = blockIdx.x * 64;
    const int tileM = blockIdx.y * 8 + wave;

    const __bf16* __restrict__ arow = A + (size_t)(tileM * 16 + lidx) * K;

    // This thread's 16B slice of the 64x32 strip: row = t>>2, chunk = t&3.
    // LDS byte offset row*64 + chunk*16 == t*16.
    const __bf16* __restrict__ gB =
        W + (size_t)(n0 + (t >> 2)) * K + (t & 3) * 8;
    const unsigned lds_base = (unsigned)(unsigned long long)(&bsm[0]);
    const unsigned myslot   = (unsigned)t * 16u;

    v8f acc[4];
#pragma unroll
    for (int nt = 0; nt < 4; ++nt) acc[nt] = v8f{};

    // prologue: DMA tile 0 into buffer 0
    {
        unsigned la = lds_base + myslot;
        asm volatile("global_load_async_to_lds_b128 %0, %1, off"
                     :: "v"(la), "v"(gB) : "memory");
    }

    const int nK = K >> 5;
    for (int kt = 0; kt < nK; ++kt) {
        asm volatile("s_wait_asynccnt 0x0" ::: "memory");
        __syncthreads();                          // strip kt landed for all waves

        if (kt + 1 < nK) {                        // overlap DMA of strip kt+1
            unsigned la = lds_base + (unsigned)((kt + 1) & 1) * 4096u + myslot;
            const __bf16* g = gB + (size_t)(kt + 1) * 32;
            asm volatile("global_load_async_to_lds_b128 %0, %1, off"
                         :: "v"(la), "v"(g) : "memory");
        }

        // A fragment from global (unique per wave, reused for 4 WMMAs)
        const int ka = kt * 32 + hi * 8;
        v8bf a0 = *(const v8bf*)(arow + ka);
        v8bf a1 = *(const v8bf*)(arow + ka + 16);
        v16bf afrag;
#pragma unroll
        for (int i = 0; i < 8; ++i) { afrag[i] = a0[i]; afrag[i + 8] = a1[i]; }
        if (kt + 1 < nK) __builtin_prefetch(arow + ka + 32, 0, 3);

        const __bf16* bbuf = &bsm[(kt & 1) * 2048];
#pragma unroll
        for (int nt = 0; nt < 4; ++nt) {
            const __bf16* bp = bbuf + (nt * 16 + lidx) * 32 + hi * 8;
            v8bf b0 = *(const v8bf*)(bp);
            v8bf b1 = *(const v8bf*)(bp + 16);
            v16bf bfrag;
#pragma unroll
            for (int i = 0; i < 8; ++i) { bfrag[i] = b0[i]; bfrag[i + 8] = b1[i]; }
            acc[nt] = __builtin_amdgcn_wmma_f32_16x16x32_bf16(
                false, afrag, false, bfrag, (short)0, acc[nt], false, false);
        }
        // next iteration's DMA targets the other buffer; reads of this buffer
        // completed before each wave reaches the barrier above -> no extra sync
    }

#pragma unroll
    for (int nt = 0; nt < 4; ++nt) {
        const int nbase = n0 + nt * 16;
        if (nbase >= N) continue;                 // wave-uniform guard
#pragma unroll
        for (int r = 0; r < 8; ++r) {
            const int mm = tileM * 16 + r + hi * 8;
            const size_t off = (size_t)mm * N + nbase + lidx;
            if (Cf) Cf[off] = acc[nt][r];
            if (Cb) Cb[off] = (__bf16)acc[nt][r];
        }
    }
}

// ---------------------------------------------------------------------------
// input (B,CIN,L) f32  ->  x_bf (B,L,CIN) bf16   (transpose + downconvert)
// ---------------------------------------------------------------------------
__global__ void transpose_to_bf16(const float* __restrict__ in,
                                  __bf16* __restrict__ x_bf)
{
    int tid = blockIdx.x * blockDim.x + threadIdx.x;     // over B*L*CIN
    if (tid >= cfg::ML * cfg::CIN) return;
    int c  = tid & (cfg::CIN - 1);
    int bl = tid >> 8;
    int l  = bl & (cfg::L - 1);
    int b  = bl >> 11;
    x_bf[tid] = (__bf16)in[((size_t)b * cfg::CIN + c) * cfg::L + l];
}

// f32 -> bf16 (weights)
__global__ void cvt_bf16(const float* __restrict__ s, __bf16* __restrict__ d, int n)
{
    int tid = blockIdx.x * blockDim.x + threadIdx.x;
    if (tid < n) d[tid] = (__bf16)s[tid];
}

// dt_proj_w (DIN,16) f32 -> (DIN,32) bf16 zero-padded in K
__global__ void cvt_dtw_pad(const float* __restrict__ s, __bf16* __restrict__ d)
{
    int tid = blockIdx.x * blockDim.x + threadIdx.x;     // over DIN*32
    if (tid >= cfg::DIN * cfg::DTK) return;
    int r = tid & (cfg::DTK - 1);
    int k = tid >> 5;
    d[tid] = (r < cfg::DTRANK) ? (__bf16)s[k * cfg::DTRANK + r] : (__bf16)0.0f;
}

// ---------------------------------------------------------------------------
// Depthwise causal conv (DCONV=4) + bias + SiLU over xr = xz[...,:DIN].
// xz layout (B,L,2*DIN) f32. Writes xc f32 and xc_bf for the x_proj GEMM.
// ---------------------------------------------------------------------------
__global__ void conv_silu(const float* __restrict__ xz,
                          const float* __restrict__ cw,   // (DIN,DCONV)
                          const float* __restrict__ cb,   // (DIN)
                          float*  __restrict__ xc,
                          __bf16* __restrict__ xc_bf)
{
    int tid = blockIdx.x * blockDim.x + threadIdx.x;     // over B*L*DIN
    if (tid >= cfg::ML * cfg::DIN) return;
    int d  = tid & (cfg::DIN - 1);
    int bl = tid >> 9;
    int l  = bl & (cfg::L - 1);
    int b  = bl >> 11;
    float acc = cb[d];
#pragma unroll
    for (int j = 0; j < cfg::DCONV; ++j) {
        int ls = l - (cfg::DCONV - 1) + j;
        if (ls >= 0)
            acc += cw[d * cfg::DCONV + j] *
                   xz[((size_t)(b * cfg::L + ls)) * (2 * cfg::DIN) + d];
    }
    float s = acc / (1.0f + __expf(-acc));               // SiLU
    xc[tid]    = s;
    xc_bf[tid] = (__bf16)s;
}

// x_db (ML,48) f32 -> dt_bf (ML,32) bf16, K zero-padded past DTRANK
__global__ void dt_pad(const float* __restrict__ xdb, __bf16* __restrict__ dt_bf)
{
    int tid = blockIdx.x * blockDim.x + threadIdx.x;     // over ML*32
    if (tid >= cfg::ML * cfg::DTK) return;
    int r  = tid & (cfg::DTK - 1);
    int bl = tid >> 5;
    dt_bf[tid] = (r < cfg::DTRANK)
               ? (__bf16)xdb[(size_t)bl * 48 + r] : (__bf16)0.0f;
}

// ---------------------------------------------------------------------------
// Selective scan: one thread per (b,d) channel, 16 states in registers,
// sequential over L. Fuses softplus(delta), the recurrence, D skip, and the
// z-gate SiLU; emits y in bf16 ready for the out_proj WMMA GEMM.
// ---------------------------------------------------------------------------
__global__ void selective_scan(const float* __restrict__ dtp,   // (ML,DIN) pre-softplus
                               const float* __restrict__ dtb,   // (DIN)
                               const float* __restrict__ xdb,   // (ML,48): [dt|B|C]
                               const float* __restrict__ xc,    // (ML,DIN)
                               const float* __restrict__ xz,    // (ML,2*DIN): z in top half
                               const float* __restrict__ A_log, // (DIN,DSTATE)
                               const float* __restrict__ Dp,    // (DIN)
                               __bf16* __restrict__ y_bf)       // (ML,DIN)
{
    int gid = blockIdx.x * blockDim.x + threadIdx.x;     // over B*DIN
    if (gid >= cfg::B * cfg::DIN) return;
    const int d = gid & (cfg::DIN - 1);
    const int b = gid >> 9;

    float Av[cfg::DSTATE];
#pragma unroll
    for (int s = 0; s < cfg::DSTATE; ++s)
        Av[s] = -__expf(A_log[d * cfg::DSTATE + s]);
    const float Dd = Dp[d], bias = dtb[d];

    float h[cfg::DSTATE];
#pragma unroll
    for (int s = 0; s < cfg::DSTATE; ++s) h[s] = 0.0f;

    for (int l = 0; l < cfg::L; ++l) {
        const size_t idx = (size_t)(b * cfg::L + l);
        float pre   = dtp[idx * cfg::DIN + d] + bias;
        float delta = (pre > 20.0f) ? pre : __logf(1.0f + __expf(pre));
        float xt = xc[idx * cfg::DIN + d];
        float dx = delta * xt;
        const float* Bm = xdb + idx * 48 + cfg::DTRANK;
        const float* Cm = Bm + cfg::DSTATE;
        float y = 0.0f;
#pragma unroll
        for (int s = 0; s < cfg::DSTATE; ++s) {
            float dA = __expf(delta * Av[s]);
            h[s] = dA * h[s] + dx * Bm[s];
            y   += h[s] * Cm[s];
        }
        float zv = xz[idx * (2 * cfg::DIN) + cfg::DIN + d];
        float o  = (y + xt * Dd) * (zv / (1.0f + __expf(-zv)));
        y_bf[idx * cfg::DIN + d] = (__bf16)o;
    }
}

// ---------------------------------------------------------------------------
// out_lin epilogue: bias + ReLU + permute (B,L,2*CIN) -> (NSPK,B,CIN,L)
// ---------------------------------------------------------------------------
__global__ void out_epilogue(const float* __restrict__ lin,   // (ML,2*CIN)
                             const float* __restrict__ bias,  // (2*CIN)
                             float* __restrict__ out)
{
    int tid = blockIdx.x * blockDim.x + threadIdx.x;     // over ML*512
    if (tid >= cfg::ML * cfg::NSPK * cfg::CIN) return;
    int o  = tid & (cfg::NSPK * cfg::CIN - 1);
    int bl = tid >> 9;
    int l  = bl & (cfg::L - 1);
    int b  = bl >> 11;
    float v = lin[tid] + bias[o];
    v = v > 0.0f ? v : 0.0f;
    int spk = o >> 8, c = o & (cfg::CIN - 1);
    out[(((size_t)spk * cfg::B + b) * cfg::CIN + c) * cfg::L + l] = v;
}

// ---------------------------------------------------------------------------
// Host orchestration
// ---------------------------------------------------------------------------
static inline int ceil_div(int a, int b) { return (a + b - 1) / b; }

extern "C" void kernel_launch(void* const* d_in, const int* in_sizes, int n_in,
                              void* d_out, int out_size, void* d_ws, size_t ws_size,
                              hipStream_t stream)
{
    using namespace cfg;
    const float* input     = (const float*)d_in[0];
    const float* in_proj_w = (const float*)d_in[1];   // (2, 1024, 256)
    const float* conv_w    = (const float*)d_in[2];   // (2, 512, 1, 4)
    const float* conv_b    = (const float*)d_in[3];   // (2, 512)
    const float* x_proj_w  = (const float*)d_in[4];   // (2, 48, 512)
    const float* dt_proj_w = (const float*)d_in[5];   // (2, 512, 16)
    const float* dt_proj_b = (const float*)d_in[6];   // (2, 512)
    const float* A_log     = (const float*)d_in[7];   // (2, 512, 16)
    const float* Dp        = (const float*)d_in[8];   // (2, 512)
    const float* out_proj_w= (const float*)d_in[9];   // (2, 256, 512)
    const float* out_lin_w = (const float*)d_in[10];  // (512, 256)
    const float* out_lin_b = (const float*)d_in[11];  // (512)
    float* out = (float*)d_out;

    // --- workspace carve-out (256B aligned) ---
    char* ws = (char*)d_ws; size_t off = 0;
    auto alloc = [&](size_t bytes) -> void* {
        void* p = ws + off; off = (off + bytes + 255) & ~(size_t)255; return p;
    };
    __bf16* x_bf    = (__bf16*)alloc((size_t)ML * CIN      * 2);  // GEMM A: in_proj / out_lin
    float*  xz      = (float*) alloc((size_t)ML * 2 * DIN  * 4);
    float*  xc      = (float*) alloc((size_t)ML * DIN      * 4);
    __bf16* xc_bf   = (__bf16*)alloc((size_t)ML * DIN      * 2);
    float*  xdb     = (float*) alloc((size_t)ML * 48       * 4);
    __bf16* dt_bf   = (__bf16*)alloc((size_t)ML * DTK      * 2);
    float*  dtp     = (float*) alloc((size_t)ML * DIN      * 4);
    __bf16* y_bf    = (__bf16*)alloc((size_t)ML * DIN      * 2);
    float*  lin     = (float*) alloc((size_t)ML * 2 * CIN  * 4);
    __bf16* w_in    = (__bf16*)alloc((size_t)2 * DIN * CIN * 2);  // 1024*256
    __bf16* w_x     = (__bf16*)alloc((size_t)64 * DIN     * 2);   // 48 rows + OOB slack
    __bf16* w_dt    = (__bf16*)alloc((size_t)DIN * DTK     * 2);
    __bf16* w_out   = (__bf16*)alloc((size_t)CIN * DIN     * 2);
    __bf16* w_lin   = (__bf16*)alloc((size_t)2 * CIN * CIN * 2);  // 512*256
    (void)ws_size; (void)in_sizes; (void)n_in; (void)out_size;

    const int T = 256;
    dim3 blk(T);

    // input -> bf16 (B,L,CIN)
    transpose_to_bf16<<<ceil_div(ML * CIN, T), blk, 0, stream>>>(input, x_bf);
    // out_lin weight (layer-independent)
    cvt_bf16<<<ceil_div(2 * CIN * CIN, T), blk, 0, stream>>>(out_lin_w, w_lin, 2 * CIN * CIN);

    for (int layer = 0; layer < 2; ++layer) {
        const float* ipw = in_proj_w + (size_t)layer * (2 * DIN) * CIN;
        const float* cw  = conv_w    + (size_t)layer * DIN * DCONV;
        const float* cb  = conv_b    + (size_t)layer * DIN;
        const float* xpw = x_proj_w  + (size_t)layer * 48 * DIN;
        const float* dpw = dt_proj_w + (size_t)layer * DIN * DTRANK;
        const float* dpb = dt_proj_b + (size_t)layer * DIN;
        const float* alg = A_log     + (size_t)layer * DIN * DSTATE;
        const float* dsk = Dp        + (size_t)layer * DIN;
        const float* opw = out_proj_w+ (size_t)layer * CIN * DIN;

        // weights -> bf16
        cvt_bf16<<<ceil_div(2 * DIN * CIN, T), blk, 0, stream>>>(ipw, w_in, 2 * DIN * CIN);
        cvt_bf16<<<ceil_div(48 * DIN, T),      blk, 0, stream>>>(xpw, w_x, 48 * DIN);
        cvt_dtw_pad<<<ceil_div(DIN * DTK, T),  blk, 0, stream>>>(dpw, w_dt);
        cvt_bf16<<<ceil_div(CIN * DIN, T),     blk, 0, stream>>>(opw, w_out, CIN * DIN);

        // 1) in_proj: (ML,256) x (1024,256)^T -> xz f32
        {
            dim3 g(ceil_div(2 * DIN, 64), ML / 128);
            wmma_gemm_bf16<<<g, dim3(256), 0, stream>>>(x_bf, w_in, xz, nullptr,
                                                        2 * DIN, CIN);
        }
        // 2) depthwise conv + SiLU
        conv_silu<<<ceil_div(ML * DIN, T), blk, 0, stream>>>(xz, cw, cb, xc, xc_bf);
        // 3) x_proj: (ML,512) x (48,512)^T -> xdb f32
        {
            dim3 g(ceil_div(48, 64), ML / 128);
            wmma_gemm_bf16<<<g, dim3(256), 0, stream>>>(xc_bf, w_x, xdb, nullptr,
                                                        48, DIN);
        }
        // 4) dt slice -> bf16 padded K=32
        dt_pad<<<ceil_div(ML * DTK, T), blk, 0, stream>>>(xdb, dt_bf);
        // 5) dt_proj: (ML,32) x (512,32)^T -> dtp f32 (pre-softplus)
        {
            dim3 g(ceil_div(DIN, 64), ML / 128);
            wmma_gemm_bf16<<<g, dim3(256), 0, stream>>>(dt_bf, w_dt, dtp, nullptr,
                                                        DIN, DTK);
        }
        // 6) selective scan (+D skip, +z gate) -> y_bf
        selective_scan<<<ceil_div(B * DIN, T), blk, 0, stream>>>(
            dtp, dpb, xdb, xc, xz, alg, dsk, y_bf);
        // 7) out_proj: (ML,512) x (256,512)^T -> bf16 directly into x_bf
        //    (feeds next layer's in_proj, or the final out_lin)
        {
            dim3 g(ceil_div(CIN, 64), ML / 128);
            wmma_gemm_bf16<<<g, dim3(256), 0, stream>>>(y_bf, w_out, nullptr, x_bf,
                                                        CIN, DIN);
        }
    }

    // out_lin: (ML,256) x (512,256)^T -> lin f32
    {
        dim3 g(ceil_div(2 * CIN, 64), ML / 128);
        wmma_gemm_bf16<<<g, dim3(256), 0, stream>>>(x_bf, w_lin, lin, nullptr,
                                                    2 * CIN, CIN);
    }
    // bias + ReLU + permute to (NSPK,B,CIN,L)
    out_epilogue<<<ceil_div(ML * NSPK * CIN, T), blk, 0, stream>>>(lin, out_lin_b, out);
}
